// MASKGCN_39565238731021
// MI455X (gfx1250) — compile-verified, matrix operand order checked by reading, not run on hardware
//
#include <hip/hip_runtime.h>

typedef __bf16 bf16;
typedef __attribute__((ext_vector_type(16))) __bf16 v16bf;
typedef __attribute__((ext_vector_type(8)))  __bf16 v8bf;
typedef __attribute__((ext_vector_type(8)))  float  v8f;

#define BATCH 512
#define NNODE 200
#define NP    224   // padded node dim: multiple of 32 (K and M-block)
#define INC   200
#define HID   256
#define OUTC  256

// ---- bf16 as a pure storage type (no reliance on __bf16 arithmetic) ----
__device__ __forceinline__ bf16 f2bf(float f) {
    union { float f; unsigned u; } in; in.f = f;
    unsigned u = in.u;
    u += 0x7FFFu + ((u >> 16) & 1u);           // round-to-nearest-even
    union { unsigned short s; bf16 b; } out; out.s = (unsigned short)(u >> 16);
    return out.b;
}
__device__ __forceinline__ float bf2f(bf16 b) {
    union { bf16 b; unsigned short s; } in; in.b = b;
    union { unsigned u; float f; } out; out.u = ((unsigned)in.s) << 16;
    return out.f;
}

// ---- prep: adj_masked = sigmoid(adj) * ((raw+raw^T)/2 + I), bf16, padded ----
__global__ void prep_adj(const float* __restrict__ adj, const float* __restrict__ raw,
                         bf16* __restrict__ Abf) {
    long long idx = (long long)blockIdx.x * blockDim.x + threadIdx.x;
    if (idx >= (long long)BATCH * NP * NP) return;
    int j = (int)(idx % NP);
    long long t = idx / NP;
    int i = (int)(t % NP);
    int b = (int)(t / NP);
    float v = 0.f;
    if (i < NNODE && j < NNODE) {
        float m = 0.5f * (raw[i * NNODE + j] + raw[j * NNODE + i]) + (i == j ? 1.f : 0.f);
        float a = adj[((long long)b * NNODE + i) * NNODE + j];
        v = m / (1.f + expf(-a));              // sigmoid(a) * m
    }
    Abf[idx] = f2bf(v);
}

// ---- prep: features -> bf16 padded [B][NP][NP] (cols = IN_C padded) ----
__global__ void prep_feat(const float* __restrict__ feat, bf16* __restrict__ Xbf) {
    long long idx = (long long)blockIdx.x * blockDim.x + threadIdx.x;
    if (idx >= (long long)BATCH * NP * NP) return;
    int j = (int)(idx % NP);
    long long t = idx / NP;
    int i = (int)(t % NP);
    int b = (int)(t / NP);
    float v = 0.f;
    if (i < NNODE && j < INC)
        v = feat[((long long)b * NNODE + i) * INC + j];
    Xbf[idx] = f2bf(v);
}

// ---- prep: W [K][Nc] f32 -> Wt [Nc][Kp] bf16 (transposed, K padded) ----
__global__ void prep_wT(const float* __restrict__ W, bf16* __restrict__ Wt,
                        int K, int Nc, int Kp) {
    int idx = blockIdx.x * blockDim.x + threadIdx.x;
    if (idx >= Nc * Kp) return;
    int k = idx % Kp, n = idx / Kp;
    Wt[idx] = f2bf(k < K ? W[k * Nc + n] : 0.f);
}

// ---- fragment load: 16x32 bf16 tile, row-major src, per-ISA A-frag layout ----
// lane l (half=l>>4, r=l&15) holds row r0+r, K = k0 + {half*8..half*8+7, 16+half*8..}
__device__ __forceinline__ v16bf load_frag(const bf16* __restrict__ base, int ld,
                                           int r0, int k0, int lane) {
    const int half = lane >> 4, r = lane & 15;
    const bf16* p = base + (long long)(r0 + r) * ld + k0 + half * 8;
    union { v16bf v; v8bf h[2]; } f;
    f.h[0] = *(const v8bf*)p;          // K = k0 + half*8 .. +7
    f.h[1] = *(const v8bf*)(p + 16);   // K = k0 + 16 + half*8 .. +7
    return f.v;
}

// ---- batched bf16 GEMM: C[b] = A[b](Mp x Kp) @ B[b]^T-stored(Np x Kp) ----
// Each wave owns a 32x64 output block: 2 M-tiles x 4 N-tiles = 8 WMMAs / K-step.
// WMMA order is j-major so both A fragments stay live from the first WMMA,
// letting the scheduler issue all 12 loads up-front with partial loadcnt waits.
// out normal [Mp][Np] (outT=0) or transposed [Np][Mp] (outT=1)
__global__ void __launch_bounds__(128)
gemm_bf16_wmma(const bf16* __restrict__ A,  long long sA,
               const bf16* __restrict__ Bt, long long sB,
               bf16* __restrict__ C,        long long sC,
               int Mp, int Kp, int Np, int outT) {
    const int lane = threadIdx.x & 31;
    const int wave = blockIdx.x * (blockDim.x >> 5) + (threadIdx.x >> 5);
    const int ntN = Np >> 6;                   // 64-col tiles
    const int ntM = Mp >> 5;                   // 32-row blocks
    const int nt = wave % ntN;
    const int mt = (wave / ntN) % ntM;
    const int b  = wave / (ntN * ntM);
    const bf16* Ab = A  + (long long)b * sA;
    const bf16* Bb = Bt + (long long)b * sB;
    bf16*       Cb = C  + (long long)b * sC;
    const int m0 = mt << 5, n0 = nt << 6;
    const int half = lane >> 4, lr = lane & 15;

    v8f acc[2][4] = {};
    for (int k0 = 0; k0 < Kp; k0 += 32) {
        // uniform-condition prefetch one K-step ahead (global_prefetch_b8)
        if (k0 + 32 < Kp) {
            __builtin_prefetch(Ab + (long long)(m0 + lr) * Kp + k0 + 32, 0, 1);
            __builtin_prefetch(Bb + (long long)(n0 + lr) * Kp + k0 + 32, 0, 1);
        }
        v16bf a0 = load_frag(Ab, Kp, m0,      k0, lane);
        v16bf a1 = load_frag(Ab, Kp, m0 + 16, k0, lane);
        v16bf b0 = load_frag(Bb, Kp, n0,      k0, lane);
        v16bf b1 = load_frag(Bb, Kp, n0 + 16, k0, lane);
        v16bf b2 = load_frag(Bb, Kp, n0 + 32, k0, lane);
        v16bf b3 = load_frag(Bb, Kp, n0 + 48, k0, lane);
        // j-major: both a0/a1 consumed immediately -> no register recycling stall
        acc[0][0] = __builtin_amdgcn_wmma_f32_16x16x32_bf16(false, a0, false, b0, (short)0, acc[0][0], false, false);
        acc[1][0] = __builtin_amdgcn_wmma_f32_16x16x32_bf16(false, a1, false, b0, (short)0, acc[1][0], false, false);
        acc[0][1] = __builtin_amdgcn_wmma_f32_16x16x32_bf16(false, a0, false, b1, (short)0, acc[0][1], false, false);
        acc[1][1] = __builtin_amdgcn_wmma_f32_16x16x32_bf16(false, a1, false, b1, (short)0, acc[1][1], false, false);
        acc[0][2] = __builtin_amdgcn_wmma_f32_16x16x32_bf16(false, a0, false, b2, (short)0, acc[0][2], false, false);
        acc[1][2] = __builtin_amdgcn_wmma_f32_16x16x32_bf16(false, a1, false, b2, (short)0, acc[1][2], false, false);
        acc[0][3] = __builtin_amdgcn_wmma_f32_16x16x32_bf16(false, a0, false, b3, (short)0, acc[0][3], false, false);
        acc[1][3] = __builtin_amdgcn_wmma_f32_16x16x32_bf16(false, a1, false, b3, (short)0, acc[1][3], false, false);
    }

    if (outT) {
        // lane's column c = n0+j*16+lr; rows m0+mi*16+half*8..+7 contiguous -> 16B stores
        #pragma unroll
        for (int mi = 0; mi < 2; ++mi)
            #pragma unroll
            for (int j = 0; j < 4; ++j) {
                v8bf pk;
                #pragma unroll
                for (int v = 0; v < 8; ++v) pk[v] = f2bf(acc[mi][j][v]);
                *(v8bf*)(Cb + (long long)(n0 + j * 16 + lr) * Mp + m0 + mi * 16 + half * 8) = pk;
            }
    } else {
        #pragma unroll
        for (int mi = 0; mi < 2; ++mi)
            #pragma unroll
            for (int j = 0; j < 4; ++j)
                #pragma unroll
                for (int v = 0; v < 8; ++v)
                    Cb[(long long)(m0 + mi * 16 + v + half * 8) * Np + n0 + j * 16 + lr] = f2bf(acc[mi][j][v]);
    }
}

// ---- readout: mean over nodes then @ pw + pb ----
__global__ void __launch_bounds__(256)
readout(const bf16* __restrict__ H, const float* __restrict__ pw,
        const float* __restrict__ pb, float* __restrict__ out) {
    const int b = blockIdx.x, c = threadIdx.x;
    const bf16* Hb = H + (long long)b * NP * OUTC;
    float s = 0.f;
    for (int r = 0; r < NNODE; ++r) s += bf2f(Hb[r * OUTC + c]);
    s *= (1.0f / NNODE);
    __shared__ float r0[256], r1[256];
    r0[c] = s * pw[c * 2 + 0];
    r1[c] = s * pw[c * 2 + 1];
    __syncthreads();
    for (int off = 128; off > 0; off >>= 1) {
        if (c < off) { r0[c] += r0[c + off]; r1[c] += r1[c + off]; }
        __syncthreads();
    }
    if (c == 0) {
        out[b * 2 + 0] = r0[0] + pb[0];
        out[b * 2 + 1] = r1[0] + pb[1];
    }
}

extern "C" void kernel_launch(void* const* d_in, const int* in_sizes, int n_in,
                              void* d_out, int out_size, void* d_ws, size_t ws_size,
                              hipStream_t stream) {
    const float* adj  = (const float*)d_in[0];
    const float* feat = (const float*)d_in[1];
    const float* raw  = (const float*)d_in[2];
    const float* W0   = (const float*)d_in[3];
    const float* W1   = (const float*)d_in[4];
    const float* W2   = (const float*)d_in[5];
    const float* pw   = (const float*)d_in[6];
    const float* pb   = (const float*)d_in[7];
    float* out = (float*)d_out;

    // carve workspace
    char* ws = (char*)d_ws;
    size_t off = 0;
    auto carve = [&](size_t bytes) {
        void* p = ws + off;
        off = (off + bytes + 255) & ~(size_t)255;
        return p;
    };
    bf16* Abf = (bf16*)carve((size_t)BATCH * NP * NP * 2);   // masked adjacency
    bf16* Xbf = (bf16*)carve((size_t)BATCH * NP * NP * 2);   // padded features
    bf16* Hb  = (bf16*)carve((size_t)BATCH * NP * HID * 2);  // layer output [NP][HID]
    bf16* St  = (bf16*)carve((size_t)BATCH * HID * NP * 2);  // support, transposed [HID][NP]
    bf16* W0t = (bf16*)carve((size_t)HID * NP * 2);
    bf16* W1t = (bf16*)carve((size_t)HID * HID * 2);
    bf16* W2t = (bf16*)carve((size_t)OUTC * HID * 2);

    // prep
    {
        long long tot = (long long)BATCH * NP * NP;
        int blocks = (int)((tot + 255) / 256);
        prep_adj <<<blocks, 256, 0, stream>>>(adj,  raw, Abf);
        prep_feat<<<blocks, 256, 0, stream>>>(feat, Xbf);
    }
    prep_wT<<<(HID * NP  + 255) / 256, 256, 0, stream>>>(W0, W0t, INC, HID, NP);
    prep_wT<<<(HID * HID + 255) / 256, 256, 0, stream>>>(W1, W1t, HID, HID, HID);
    prep_wT<<<(HID * HID + 255) / 256, 256, 0, stream>>>(W2, W2t, HID, HID, HID);

    // 1 wave per 32x64 output block; 4 waves per block, exact grid -> EXEC all ones
    const int waves  = BATCH * (NP / 32) * (HID / 64);   // 512*7*4 = 14336
    const dim3 gblk(waves / 4), gth(128);
    const long long sAdj = (long long)NP * NP;
    const long long sH   = (long long)NP * HID;
    const long long sSt  = (long long)HID * NP;

    // layer 0: St = (X @ W0)^T ; Hb = Adj @ S
    gemm_bf16_wmma<<<gblk, gth, 0, stream>>>(Xbf, sAdj, W0t, 0LL, St, sSt, NP, NP,  HID, 1);
    gemm_bf16_wmma<<<gblk, gth, 0, stream>>>(Abf, sAdj, St,  sSt, Hb, sH,  NP, NP,  HID, 0);
    // layer 1: St = (H @ W1)^T ; Hb = Adj @ S
    gemm_bf16_wmma<<<gblk, gth, 0, stream>>>(Hb,  sH,   W1t, 0LL, St, sSt, NP, HID, HID, 1);
    gemm_bf16_wmma<<<gblk, gth, 0, stream>>>(Abf, sAdj, St,  sSt, Hb, sH,  NP, NP,  HID, 0);
    // layer 2: St = (H @ W2)^T ; Hb = Adj @ S
    gemm_bf16_wmma<<<gblk, gth, 0, stream>>>(Hb,  sH,   W2t, 0LL, St, sSt, NP, HID, HID, 1);
    gemm_bf16_wmma<<<gblk, gth, 0, stream>>>(Abf, sAdj, St,  sSt, Hb, sH,  NP, NP,  HID, 0);

    // mean over nodes, then @ pw + pb
    readout<<<BATCH, 256, 0, stream>>>(Hb, pw, pb, out);
}
